// TransformerLayer_43490838839649
// MI455X (gfx1250) — matrix operand order repaired and checked by reading, not compiled, not run on hardware
//
#include <hip/hip_runtime.h>

// ---------------- problem constants ----------------
#define NB    4
#define SEQ   1024
#define DMODEL 1024
#define NTOK  (NB*SEQ)          // 4096
#define HDIM  4096              // MULT*D
#define NHEAD 16
#define HD    64
#define NEXP  16
#define TOPK  2
#define CAP   640               // int(B*1.25*2/16) with B=4096

#if defined(__gfx1250__) && __has_builtin(__builtin_amdgcn_global_load_async_to_lds_b128)
#define HAVE_ASYNC_LDS 1
#else
#define HAVE_ASYNC_LDS 0
#endif

typedef __attribute__((ext_vector_type(16))) __bf16 v16bf;
typedef __attribute__((ext_vector_type(8)))  float  v8f;
typedef __attribute__((ext_vector_type(4)))  int    v4i;

#if HAVE_ASYNC_LDS
typedef __attribute__((address_space(1))) v4i* gv4i_p;   // global int4 vec ptr
typedef __attribute__((address_space(3))) v4i* lv4i_p;   // LDS int4 vec ptr
#endif

struct U8x4 { uint4 lo, hi; };
static_assert(sizeof(U8x4) == sizeof(v16bf), "frag size");

__device__ __forceinline__ v16bf make_frag(uint4 a, uint4 b) {
    U8x4 u; u.lo = a; u.hi = b;
    return __builtin_bit_cast(v16bf, u);
}
__device__ __forceinline__ unsigned short f2bf(float f) {
    unsigned u = __builtin_bit_cast(unsigned, f);
    unsigned r = u + 0x7FFFu + ((u >> 16) & 1u);
    return (unsigned short)(r >> 16);
}
__device__ __forceinline__ float bf2f(unsigned short h) {
    unsigned u = ((unsigned)h) << 16;
    return __builtin_bit_cast(float, u);
}
__device__ __forceinline__ float gelu_exact(float x) {
    return 0.5f * x * (1.0f + erff(x * 0.70710678118654752f));
}

// ---------------- LayerNorm: f32 in -> bf16 (and optional f32) out ----------------
__global__ __launch_bounds__(256)
void ln_kernel(const float* __restrict__ x, const float* __restrict__ g,
               const float* __restrict__ be, unsigned short* __restrict__ ob,
               float* __restrict__ of, int D) {
    const int row = blockIdx.x, tid = threadIdx.x;
    const float* xr = x + (long)row * D;
    float s = 0.f, s2 = 0.f;
    for (int j = tid; j < D; j += 256) { float v = xr[j]; s += v; s2 += v * v; }
    __shared__ float r1[256], r2[256];
    r1[tid] = s; r2[tid] = s2; __syncthreads();
    for (int st = 128; st > 0; st >>= 1) {
        if (tid < st) { r1[tid] += r1[tid + st]; r2[tid] += r2[tid + st]; }
        __syncthreads();
    }
    const float mu = r1[0] / (float)D;
    const float var = r2[0] / (float)D - mu * mu;
    const float rstd = rsqrtf(var + 1e-5f);
    for (int j = tid; j < D; j += 256) {
        float y = (xr[j] - mu) * rstd * g[j] + be[j];
        ob[(long)row * D + j] = f2bf(y);
        if (of) of[(long)row * D + j] = y;
    }
}

// ---------------- WMMA GEMM: C[M,N] = act(A[M,K](bf16) * op(B)(f32->bf16) + bias) + res ----
// TRANSB=0: B is [N,K] row-major (y = x @ W^T).  TRANSB=1: B is [K,N] row-major.
template <int TRANSB, int ACT, int OUT_BF16, int HAS_BIAS, int HAS_RES>
__global__ __launch_bounds__(256)
void gemm_wmma(const unsigned short* __restrict__ A, int lda,
               const float* __restrict__ Bm,
               const float* __restrict__ bias,
               const float* __restrict__ resid,
               float* __restrict__ Cf, unsigned short* __restrict__ Cb,
               int N, int Kdim,
               long strideA, long strideB, long strideC) {
    const int tid = threadIdx.x, lane = tid & 31, wid = tid >> 5;
    const int r = lane & 15, hi = lane >> 4;
    const int bm = blockIdx.y * 128, bn = blockIdx.x * 128;
    const int ez = blockIdx.z;
    A  += (long)ez * strideA;
    Bm += (long)ez * strideB;

    __shared__ alignas(16) unsigned short sA[128][32];
    __shared__ alignas(16) unsigned short sB[128][32];   // stored as [n][k]

    v8f acc[2][4];
#pragma unroll
    for (int i = 0; i < 2; i++)
#pragma unroll
        for (int j = 0; j < 4; j++) acc[i][j] = v8f{0.f,0.f,0.f,0.f,0.f,0.f,0.f,0.f};

    const int wm = (wid & 3) * 32;   // 4 waves tile M
    const int wn = (wid >> 2) * 64;  // 2 waves tile N

    for (int kt = 0; kt < Kdim; kt += 32) {
        __syncthreads();
        {   // stage A tile 128x32 bf16 (32B per thread) -- async DMA to LDS on CDNA5
            const int row = tid >> 1, kc = (tid & 1) * 16;
            const unsigned short* src = A + (long)(bm + row) * lda + kt + kc;
#if HAVE_ASYNC_LDS
            __builtin_amdgcn_global_load_async_to_lds_b128(
                (gv4i_p)(void*)src, (lv4i_p)(void*)&sA[row][kc], 0, 0);
            __builtin_amdgcn_global_load_async_to_lds_b128(
                (gv4i_p)(void*)src, (lv4i_p)(void*)&sA[row][kc], 16, 0);
#else
            uint4 c0 = *(const uint4*)src;
            uint4 c1 = *(const uint4*)(src + 8);
            *(uint4*)&sA[row][kc]     = c0;
            *(uint4*)&sA[row][kc + 8] = c1;
#endif
            if (kt + 32 < Kdim) __builtin_prefetch(src + 32, 0, 1);
        }
        if (TRANSB == 0) {           // B[N,K]: rows contiguous in k (convert f32->bf16)
            const int row = tid >> 1, kc = (tid & 1) * 16;
            const float* src = Bm + (long)(bn + row) * Kdim + kt + kc;
            unsigned short tmp[16];
#pragma unroll
            for (int j = 0; j < 16; j++) tmp[j] = f2bf(src[j]);
            *(uint4*)&sB[row][kc]     = *(const uint4*)&tmp[0];
            *(uint4*)&sB[row][kc + 8] = *(const uint4*)&tmp[8];
            if (kt + 32 < Kdim) __builtin_prefetch(src + 32, 0, 1);
        } else {                     // B[K,N]: coalesce over n, transpose into LDS
            const int krow = tid >> 3, nc = (tid & 7) * 16;
            const float* src = Bm + (long)(kt + krow) * N + bn + nc;
#pragma unroll
            for (int j = 0; j < 16; j++) sB[nc + j][krow] = f2bf(src[j]);
            if (kt + 32 < Kdim) __builtin_prefetch(src + (long)32 * N, 0, 1);
        }
#if HAVE_ASYNC_LDS
        asm volatile("s_wait_asynccnt 0" ::: "memory");
#endif
        __syncthreads();

        v16bf af[2];
#pragma unroll
        for (int mf = 0; mf < 2; mf++) {
            const int row = wm + mf * 16 + r;
            uint4 c0 = *(const uint4*)&sA[row][hi * 8];
            uint4 c1 = *(const uint4*)&sA[row][16 + hi * 8];
            af[mf] = make_frag(c0, c1);
        }
#pragma unroll
        for (int nf = 0; nf < 4; nf++) {
            const int col = wn + nf * 16 + r;
            uint4 c0 = *(const uint4*)&sB[col][hi * 8];
            uint4 c1 = *(const uint4*)&sB[col][16 + hi * 8];
            v16bf bfr = make_frag(c0, c1);
#pragma unroll
            for (int mf = 0; mf < 2; mf++)
                acc[mf][nf] = __builtin_amdgcn_wmma_f32_16x16x32_bf16(
                    false, af[mf], false, bfr, (short)0, acc[mf][nf], false, false);
        }
    }

    // epilogue
#pragma unroll
    for (int mf = 0; mf < 2; mf++)
#pragma unroll
        for (int nf = 0; nf < 4; nf++) {
            const int col = bn + wn + nf * 16 + r;
            float bv = HAS_BIAS ? bias[col] : 0.f;
#pragma unroll
            for (int i = 0; i < 8; i++) {
                const int row = bm + wm + mf * 16 + hi * 8 + i;
                float v = acc[mf][nf][i] + bv;
                if (ACT == 1) v = gelu_exact(v);
                if (HAS_RES) v += resid[(long)row * N + col];
                const long o = (long)ez * strideC + (long)row * N + col;
                if (OUT_BF16) Cb[o] = f2bf(v);
                else          Cf[o] = v;
            }
        }
}

// ---------------- repack q,k -> [b,h,s,hd], v -> [b,h,hd,s] ----------------
__global__ __launch_bounds__(256)
void repack_qkv(const unsigned short* __restrict__ qi,
                const unsigned short* __restrict__ ki,
                const unsigned short* __restrict__ vi,
                unsigned short* __restrict__ qh,
                unsigned short* __restrict__ kh,
                unsigned short* __restrict__ vt) {
    const long idx = (long)blockIdx.x * 256 + threadIdx.x;   // over NTOK*DMODEL
    const int t = (int)(idx >> 10), f = (int)(idx & 1023);
    const int b = t >> 10, s = t & 1023;
    const int h = f >> 6, d = f & 63;
    const long o1 = (((long)(b * NHEAD + h)) * SEQ + s) * HD + d;
    qh[o1] = qi[idx];
    kh[o1] = ki[idx];
    const long o2 = (((long)(b * NHEAD + h)) * HD + d) * SEQ + s;
    vt[o2] = vi[idx];
}

// ---------------- flash-style causal attention ----------------
__global__ __launch_bounds__(256)
void attn_kernel(const unsigned short* __restrict__ qh,
                 const unsigned short* __restrict__ kh,
                 const unsigned short* __restrict__ vt,
                 unsigned short* __restrict__ aob) {
    const int lane = threadIdx.x & 31, wid = threadIdx.x >> 5;
    const int r = lane & 15, hi = lane >> 4;
    const int h = blockIdx.y, b = blockIdx.z;
    const int q0 = blockIdx.x * 128 + wid * 16;
    const float NEG = -1e30f;

    __shared__ alignas(16) unsigned short pbuf[8][16][32];

    const unsigned short* qp = qh + (((long)(b * NHEAD + h)) * SEQ + q0 + r) * HD;
    v16bf aq[2];
#pragma unroll
    for (int j = 0; j < 2; j++) {
        uint4 c0 = *(const uint4*)(qp + j * 32 + hi * 16);
        uint4 c1 = *(const uint4*)(qp + j * 32 + hi * 16 + 8);
        aq[j] = make_frag(c0, c1);
    }

    v8f accO[4];
#pragma unroll
    for (int f = 0; f < 4; f++) accO[f] = v8f{0.f,0.f,0.f,0.f,0.f,0.f,0.f,0.f};
    float mrow[8], lrow[8];
#pragma unroll
    for (int i = 0; i < 8; i++) { mrow[i] = NEG; lrow[i] = 0.f; }

    const unsigned short* kb = kh + ((long)(b * NHEAD + h)) * SEQ * HD;
    const unsigned short* vb = vt + ((long)(b * NHEAD + h)) * HD * SEQ;
    const int ktmax = (q0 + 15) >> 5;

    for (int kt = 0; kt <= ktmax; kt++) {
        v8f s0 = v8f{0.f,0.f,0.f,0.f,0.f,0.f,0.f,0.f};
        v8f s1 = s0;
#pragma unroll
        for (int j = 0; j < 2; j++) {
            {   const int col = kt * 32 + r;
                const unsigned short* kp = kb + (long)col * HD + j * 32 + hi * 16;
                v16bf bk = make_frag(*(const uint4*)kp, *(const uint4*)(kp + 8));
                s0 = __builtin_amdgcn_wmma_f32_16x16x32_bf16(false, aq[j], false, bk,
                                                             (short)0, s0, false, false); }
            {   const int col = kt * 32 + 16 + r;
                const unsigned short* kp = kb + (long)col * HD + j * 32 + hi * 16;
                v16bf bk = make_frag(*(const uint4*)kp, *(const uint4*)(kp + 8));
                s1 = __builtin_amdgcn_wmma_f32_16x16x32_bf16(false, aq[j], false, bk,
                                                             (short)0, s1, false, false); }
        }
        float fac[8], p0[8], p1[8];
#pragma unroll
        for (int i = 0; i < 8; i++) {
            const int row = q0 + hi * 8 + i;
            const int key0 = kt * 32 + r, key1 = key0 + 16;
            float a = s0[i] * 0.125f; if (key0 > row) a = NEG;
            float c = s1[i] * 0.125f; if (key1 > row) c = NEG;
            float mx = fmaxf(a, c);
#pragma unroll
            for (int m = 1; m < 16; m <<= 1) mx = fmaxf(mx, __shfl_xor(mx, m, 32));
            const float mnew = fmaxf(mrow[i], mx);
            const bool dead = (mnew < -1e29f);
            float f = dead ? 0.f : __expf(mrow[i] - mnew);
            float e0 = dead ? 0.f : __expf(a - mnew);
            float e1 = dead ? 0.f : __expf(c - mnew);
            float rs = e0 + e1;
#pragma unroll
            for (int m = 1; m < 16; m <<= 1) rs += __shfl_xor(rs, m, 32);
            lrow[i] = lrow[i] * f + rs;
            mrow[i] = mnew;
            fac[i] = f; p0[i] = e0; p1[i] = e1;
        }
#pragma unroll
        for (int f4 = 0; f4 < 4; f4++)
#pragma unroll
            for (int i = 0; i < 8; i++) accO[f4][i] *= fac[i];

        // C-layout -> A-layout transpose bounce through per-wave LDS
#pragma unroll
        for (int i = 0; i < 8; i++) {
            pbuf[wid][hi * 8 + i][r]      = f2bf(p0[i]);
            pbuf[wid][hi * 8 + i][16 + r] = f2bf(p1[i]);
        }
        asm volatile("s_wait_dscnt 0" ::: "memory");
        v16bf pa = make_frag(*(const uint4*)&pbuf[wid][r][hi * 16],
                             *(const uint4*)&pbuf[wid][r][hi * 16 + 8]);
#pragma unroll
        for (int f4 = 0; f4 < 4; f4++) {
            const int d = f4 * 16 + r;
            const unsigned short* vp = vb + (long)d * SEQ + kt * 32 + hi * 16;
            v16bf bv = make_frag(*(const uint4*)vp, *(const uint4*)(vp + 8));
            accO[f4] = __builtin_amdgcn_wmma_f32_16x16x32_bf16(false, pa, false, bv,
                                                               (short)0, accO[f4], false, false);
        }
    }
#pragma unroll
    for (int i = 0; i < 8; i++) {
        const float inv = 1.0f / lrow[i];
        const int row = q0 + hi * 8 + i;
#pragma unroll
        for (int f4 = 0; f4 < 4; f4++) {
            const int col = h * HD + f4 * 16 + r;
            aob[((long)b * SEQ + row) * DMODEL + col] = f2bf(accO[f4][i] * inv);
        }
    }
}

// ---------------- router logits: [NTOK,H]bf16 @ r_w2[16,H]^T + b2 ----------------
__global__ __launch_bounds__(256)
void router_logits_kernel(const unsigned short* __restrict__ rh,
                          const float* __restrict__ w2, const float* __restrict__ b2,
                          float* __restrict__ lg_ws, float* __restrict__ lg_out) {
    const int t = blockIdx.x, tid = threadIdx.x;
    const unsigned short* hr = rh + (long)t * HDIM;
    float part[NEXP];
#pragma unroll
    for (int e = 0; e < NEXP; e++) part[e] = 0.f;
    for (int j = tid; j < HDIM; j += 256) {
        const float hv = bf2f(hr[j]);
#pragma unroll
        for (int e = 0; e < NEXP; e++) part[e] += hv * w2[e * HDIM + j];
    }
    __shared__ float red[256];
    for (int e = 0; e < NEXP; e++) {
        red[tid] = part[e]; __syncthreads();
        for (int s = 128; s > 0; s >>= 1) {
            if (tid < s) red[tid] += red[tid + s];
            __syncthreads();
        }
        if (tid == 0) {
            const float v = red[0] + b2[e];
            lg_ws[t * NEXP + e] = v;
            lg_out[t * NEXP + e] = v;
        }
        __syncthreads();
    }
}

// ---------------- top-2 + softmax probs ----------------
__global__ __launch_bounds__(256)
void topk_kernel(const float* __restrict__ lg, int* __restrict__ inds,
                 float* __restrict__ probs) {
    const int t = blockIdx.x * 256 + threadIdx.x;
    if (t >= NTOK) return;
    const float* l = lg + (long)t * NEXP;
    float v1 = -3e38f, v2 = -3e38f; int i1 = 0, i2 = 0;
#pragma unroll
    for (int e = 0; e < NEXP; e++) {
        const float v = l[e];
        if (v > v1)      { v2 = v1; i2 = i1; v1 = v; i1 = e; }
        else if (v > v2) { v2 = v;  i2 = e; }
    }
    const float e2 = __expf(v2 - v1);
    const float den = 1.f + e2;
    inds[2 * t] = i1; inds[2 * t + 1] = i2;
    probs[2 * t] = 1.f / den; probs[2 * t + 1] = e2 / den;
}

// ---------------- per-expert prefix scan -> capacity slots ----------------
__global__ __launch_bounds__(1024)
void route_scan(const int* __restrict__ inds, int* __restrict__ flat) {
    const int e = blockIdx.x, tid = threadIdx.x;
    __shared__ int sc[1024];
    int base = 0;
    for (int seg = 0; seg < (NTOK * TOPK) / 1024; seg++) {
        const int i = seg * 1024 + tid;
        const int fl = (inds[i] == e) ? 1 : 0;
        sc[tid] = fl; __syncthreads();
        for (int off = 1; off < 1024; off <<= 1) {
            const int v = (tid >= off) ? sc[tid - off] : 0;
            __syncthreads();
            sc[tid] += v;
            __syncthreads();
        }
        if (fl) {
            const int pos = base + sc[tid] - 1;
            flat[i] = (pos < CAP) ? (e * CAP + pos) : -1;
        }
        base += sc[1023];
        __syncthreads();
    }
}

__global__ __launch_bounds__(256)
void zero_u16(unsigned short* __restrict__ p, long n) {
    const long i = (long)blockIdx.x * 256 + threadIdx.x;
    if (i < n) p[i] = 0;
}

__global__ __launch_bounds__(256)
void scatter_kernel(const unsigned short* __restrict__ xn2b,
                    const int* __restrict__ flat, unsigned short* __restrict__ ein) {
    const int i = blockIdx.x;
    const int f = flat[i];
    if (f < 0) return;
    const unsigned short* src = xn2b + (long)(i >> 1) * DMODEL;
    unsigned short* dst = ein + (long)f * DMODEL;
    for (int j = threadIdx.x; j < DMODEL; j += 256) dst[j] = src[j];
}

__global__ __launch_bounds__(256)
void combine_kernel(const float* __restrict__ x1, const float* __restrict__ xn2f,
                    const float* __restrict__ eout, const int* __restrict__ flat,
                    const float* __restrict__ probs, float* __restrict__ out) {
    const int t = blockIdx.x;
    const int f0 = flat[2 * t], f1 = flat[2 * t + 1];
    const float p0 = probs[2 * t], p1 = probs[2 * t + 1];
    const bool any = (f0 >= 0) || (f1 >= 0);
    for (int d = threadIdx.x; d < DMODEL; d += 256) {
        float m;
        if (any) {
            m = 0.f;
            if (f0 >= 0) m += eout[(long)f0 * DMODEL + d] * p0;
            if (f1 >= 0) m += eout[(long)f1 * DMODEL + d] * p1;
        } else {
            m = xn2f[(long)t * DMODEL + d];
        }
        out[(long)t * DMODEL + d] = x1[(long)t * DMODEL + d] + m;
    }
}

// ---------------- launch ----------------
extern "C" void kernel_launch(void* const* d_in, const int* in_sizes, int n_in,
                              void* d_out, int out_size, void* d_ws, size_t ws_size,
                              hipStream_t stream) {
    (void)in_sizes; (void)n_in; (void)out_size; (void)ws_size;
    const float* x      = (const float*)d_in[0];
    const float* ln1_g  = (const float*)d_in[1];
    const float* ln1_b  = (const float*)d_in[2];
    const float* qkv_w  = (const float*)d_in[3];
    const float* qkv_b  = (const float*)d_in[4];
    const float* ain_w  = (const float*)d_in[5];
    const float* ain_b  = (const float*)d_in[6];
    const float* aout_w = (const float*)d_in[7];
    const float* aout_b = (const float*)d_in[8];
    const float* ln2_g  = (const float*)d_in[9];
    const float* ln2_b  = (const float*)d_in[10];
    const float* r_w1   = (const float*)d_in[11];
    const float* r_b1   = (const float*)d_in[12];
    const float* r_w2   = (const float*)d_in[13];
    const float* r_b2   = (const float*)d_in[14];
    const float* mlp1   = (const float*)d_in[15];
    const float* mlp2   = (const float*)d_in[16];

    float* out = (float*)d_out;
    float* logits_out = out + (size_t)NTOK * DMODEL;

    char* ws = (char*)d_ws;
    size_t off = 0;
    auto alloc = [&](size_t bytes) -> void* {
        void* p = ws + off;
        off += (bytes + 255) & ~(size_t)255;
        return p;
    };
    unsigned short* xn_bf  = (unsigned short*)alloc((size_t)NTOK * DMODEL * 2);
    unsigned short* qkv_bf = (unsigned short*)alloc((size_t)NTOK * 3 * DMODEL * 2);
    unsigned short* qi     = (unsigned short*)alloc((size_t)NTOK * DMODEL * 2);
    unsigned short* ki     = (unsigned short*)alloc((size_t)NTOK * DMODEL * 2);
    unsigned short* vi     = (unsigned short*)alloc((size_t)NTOK * DMODEL * 2);
    unsigned short* qh     = (unsigned short*)alloc((size_t)NTOK * DMODEL * 2);
    unsigned short* kh     = (unsigned short*)alloc((size_t)NTOK * DMODEL * 2);
    unsigned short* vt     = (unsigned short*)alloc((size_t)NTOK * DMODEL * 2);
    unsigned short* ao_bf  = (unsigned short*)alloc((size_t)NTOK * DMODEL * 2);
    float*          x1     = (float*)alloc((size_t)NTOK * DMODEL * 4);
    unsigned short* xn2_bf = (unsigned short*)alloc((size_t)NTOK * DMODEL * 2);
    float*          xn2_f  = (float*)alloc((size_t)NTOK * DMODEL * 4);
    unsigned short* rh_bf  = (unsigned short*)alloc((size_t)NTOK * HDIM * 2);
    float*          lg_ws  = (float*)alloc((size_t)NTOK * NEXP * 4);
    int*            inds   = (int*)alloc((size_t)NTOK * TOPK * 4);
    float*          probs  = (float*)alloc((size_t)NTOK * TOPK * 4);
    int*            flat   = (int*)alloc((size_t)NTOK * TOPK * 4);
    unsigned short* ein    = (unsigned short*)alloc((size_t)NEXP * CAP * DMODEL * 2);
    unsigned short* mid_bf = (unsigned short*)alloc((size_t)NEXP * CAP * HDIM * 2);
    float*          eout   = (float*)alloc((size_t)NEXP * CAP * DMODEL * 4);

    // 1. LN1
    ln_kernel<<<NTOK, 256, 0, stream>>>(x, ln1_g, ln1_b, xn_bf, nullptr, DMODEL);

    // 2. QKV projection: [4096,1024] @ [3072,1024]^T -> bf16
    gemm_wmma<0,0,1,1,0><<<dim3(3 * DMODEL / 128, NTOK / 128, 1), 256, 0, stream>>>(
        xn_bf, DMODEL, qkv_w, qkv_b, nullptr, nullptr, qkv_bf, 3 * DMODEL, DMODEL, 0, 0, 0);

    // 3. attention in-projections (q0@wq^T etc.), A = slices of qkv_bf (lda=3072)
    gemm_wmma<0,0,1,1,0><<<dim3(DMODEL / 128, NTOK / 128, 1), 256, 0, stream>>>(
        qkv_bf + 0 * DMODEL, 3 * DMODEL, ain_w + (size_t)0 * DMODEL * DMODEL,
        ain_b + 0 * DMODEL, nullptr, nullptr, qi, DMODEL, DMODEL, 0, 0, 0);
    gemm_wmma<0,0,1,1,0><<<dim3(DMODEL / 128, NTOK / 128, 1), 256, 0, stream>>>(
        qkv_bf + 1 * DMODEL, 3 * DMODEL, ain_w + (size_t)1 * DMODEL * DMODEL,
        ain_b + 1 * DMODEL, nullptr, nullptr, ki, DMODEL, DMODEL, 0, 0, 0);
    gemm_wmma<0,0,1,1,0><<<dim3(DMODEL / 128, NTOK / 128, 1), 256, 0, stream>>>(
        qkv_bf + 2 * DMODEL, 3 * DMODEL, ain_w + (size_t)2 * DMODEL * DMODEL,
        ain_b + 2 * DMODEL, nullptr, nullptr, vi, DMODEL, DMODEL, 0, 0, 0);

    // 4. repack heads
    repack_qkv<<<(NTOK * DMODEL) / 256, 256, 0, stream>>>(qi, ki, vi, qh, kh, vt);

    // 5. causal attention
    attn_kernel<<<dim3(SEQ / 128, NHEAD, NB), 256, 0, stream>>>(qh, kh, vt, ao_bf);

    // 6. out projection + residual -> x1 (fp32)
    gemm_wmma<0,0,0,1,1><<<dim3(DMODEL / 128, NTOK / 128, 1), 256, 0, stream>>>(
        ao_bf, DMODEL, aout_w, aout_b, x, x1, nullptr, DMODEL, DMODEL, 0, 0, 0);

    // 7. LN2 (bf16 + fp32 copies)
    ln_kernel<<<NTOK, 256, 0, stream>>>(x1, ln2_g, ln2_b, xn2_bf, xn2_f, DMODEL);

    // 8. router hidden: gelu(xn2 @ r_w1^T + b1) -> bf16 [4096,4096]
    gemm_wmma<0,1,1,1,0><<<dim3(HDIM / 128, NTOK / 128, 1), 256, 0, stream>>>(
        xn2_bf, DMODEL, r_w1, r_b1, nullptr, nullptr, rh_bf, HDIM, DMODEL, 0, 0, 0);

    // 9. router logits (N=16, plain reduction) -> ws + d_out logits
    router_logits_kernel<<<NTOK, 256, 0, stream>>>(rh_bf, r_w2, r_b2, lg_ws, logits_out);

    // 10. top-2 + probs
    topk_kernel<<<NTOK / 256, 256, 0, stream>>>(lg_ws, inds, probs);

    // 11. capacity slot assignment (per-expert prefix scan)
    route_scan<<<NEXP, 1024, 0, stream>>>(inds, flat);

    // 12. zero + scatter expert inputs
    zero_u16<<<(int)(((size_t)NEXP * CAP * DMODEL + 255) / 256), 256, 0, stream>>>(
        ein, (long)NEXP * CAP * DMODEL);
    scatter_kernel<<<NTOK * TOPK, 256, 0, stream>>>(xn2_bf, flat, ein);

    // 13. expert GEMM 1: gelu(ein @ mlp1[e]) -> mid bf16   (B is [D,H] row-major, NN)
    gemm_wmma<1,1,1,0,0><<<dim3(HDIM / 128, CAP / 128, NEXP), 256, 0, stream>>>(
        ein, DMODEL, mlp1, nullptr, nullptr, nullptr, mid_bf, HDIM, DMODEL,
        (long)CAP * DMODEL, (long)DMODEL * HDIM, (long)CAP * HDIM);

    // 14. expert GEMM 2: mid @ mlp2[e] -> eout fp32        (B is [H,D] row-major, NN)
    gemm_wmma<1,0,0,0,0><<<dim3(DMODEL / 128, CAP / 128, NEXP), 256, 0, stream>>>(
        mid_bf, HDIM, mlp2, nullptr, nullptr, eout, nullptr, DMODEL, HDIM,
        (long)CAP * HDIM, (long)HDIM * DMODEL, (long)CAP * DMODEL);

    // 15. combine + residual -> d_out
    combine_kernel<<<NTOK, 256, 0, stream>>>(x1, xn2_f, eout, flat, probs, out);
}
